// TtCausalSelfAttention_40845138985099
// MI455X (gfx1250) — matrix operand order repaired and checked by reading, not compiled, not run on hardware
//
#include <hip/hip_runtime.h>

#define B_  4
#define T_  2048
#define C_  1024
#define H_  16
#define HD_ 64
#define M_  (B_ * T_)   // 8192
#define N3_ (3 * C_)    // 3072

typedef __attribute__((ext_vector_type(16))) _Float16 v16h;
typedef __attribute__((ext_vector_type(8)))  _Float16 v8h;
typedef __attribute__((ext_vector_type(4)))  _Float16 v4h;
typedef __attribute__((ext_vector_type(8)))  float    v8f;
typedef __attribute__((ext_vector_type(4)))  int      v4i;

union FragU { v16h v; v8h h[2]; };

// ---- gfx1250 async global->LDS copy (ASYNCcnt-tracked), with fallback ----
#if defined(__has_builtin)
#if __has_builtin(__builtin_amdgcn_global_load_async_to_lds_b128)
#define USE_ASYNC_LDS 1
#endif
#endif

typedef __attribute__((address_space(1))) v4i gv4i_t;  // global int4
typedef __attribute__((address_space(3))) v4i lv4i_t;  // LDS int4

__device__ __forceinline__ void g2l_b128(const _Float16* g, _Float16* l) {
#ifdef USE_ASYNC_LDS
  __builtin_amdgcn_global_load_async_to_lds_b128((gv4i_t*)g, (lv4i_t*)l, 0, 0);
#else
  *(v8h*)l = *(const v8h*)g;
#endif
}

__device__ __forceinline__ void wait_async() {
#ifdef USE_ASYNC_LDS
#if __has_builtin(__builtin_amdgcn_s_wait_asynccnt)
  __builtin_amdgcn_s_wait_asynccnt(0);
#else
  asm volatile("s_wait_asynccnt 0x0" ::: "memory");
#endif
#endif
}

__device__ __forceinline__ v8f wmma16(v16h a, v16h b, v8f c) {
  return __builtin_amdgcn_wmma_f32_16x16x32_f16(false, a, false, b, (short)0, c,
                                                false, false);
}

// A-fragment (16x32, MxK): lane row M=lane&15; halves 0..7 -> K=hi*8+0..7,
// halves 8..15 -> K=16+hi*8+0..7
__device__ __forceinline__ v16h load_fragA(const _Float16* tile, int stride, int lane) {
  const int row = lane & 15, hi = (lane >> 4) & 1;
  const _Float16* p = tile + row * stride + hi * 8;
  FragU f;
  f.h[0] = *(const v8h*)(p);
  f.h[1] = *(const v8h*)(p + 16);
  return f.v;
}

// B-fragment (32x16, KxN) from LDS tile stored [N][K]: lane col N=lane&15;
// halves 0..7 -> K=hi*8+0..7, halves 8..15 -> K=16+hi*8+0..7
__device__ __forceinline__ v16h load_fragB(const _Float16* tile, int stride, int lane) {
  const int row = lane & 15, hi = (lane >> 4) & 1;
  const _Float16* p = tile + row * stride + hi * 8;
  FragU f;
  f.h[0] = *(const v8h*)(p);
  f.h[1] = *(const v8h*)(p + 16);
  return f.v;
}

// ---------------------------------------------------------------------------
// Kernel 1: qkv = x @ w_attn^T + b_attn ; scatter into Q (scaled), K, V^T (f16)
// ---------------------------------------------------------------------------
__global__ __launch_bounds__(128) void qkv_gemm(const float* __restrict__ x,
                                                const float* __restrict__ w,
                                                const float* __restrict__ bias,
                                                _Float16* __restrict__ Q,
                                                _Float16* __restrict__ K,
                                                _Float16* __restrict__ Vt) {
  __shared__ __align__(16) _Float16 As[64 * 40];
  __shared__ __align__(16) _Float16 Bs[64 * 40];
  const int tid  = threadIdx.x;
  const int lane = tid & 31, wv = tid >> 5;
  const int hi = (lane >> 4) & 1, ln = lane & 15;
  const int mBase = blockIdx.x * 64;
  const int fBase = blockIdx.y * 64;

  v8f acc[4] = {};
  const int r0 = tid >> 3;        // 0..15
  const int c4 = (tid & 7) * 4;   // 0..28

  for (int kc = 0; kc < C_; kc += 32) {
    // issue all 8 global loads first so they pipeline under one wait
    float4 av[4], bv[4];
#pragma unroll
    for (int rr = 0; rr < 4; ++rr) {
      const int row = rr * 16 + r0;
      av[rr] = *(const float4*)(x + (size_t)(mBase + row) * C_ + kc + c4);
      bv[rr] = *(const float4*)(w + (size_t)(fBase + row) * C_ + kc + c4);
    }
#pragma unroll
    for (int rr = 0; rr < 4; ++rr) {
      const int row = rr * 16 + r0;
      v4h ah = {(_Float16)av[rr].x, (_Float16)av[rr].y, (_Float16)av[rr].z, (_Float16)av[rr].w};
      *(v4h*)(As + row * 40 + c4) = ah;
      v4h bh4 = {(_Float16)bv[rr].x, (_Float16)bv[rr].y, (_Float16)bv[rr].z, (_Float16)bv[rr].w};
      *(v4h*)(Bs + row * 40 + c4) = bh4;
    }
    __syncthreads();
    // preload all fragments, then back-to-back WMMAs
    const v16h aF = load_fragA(As + wv * 16 * 40, 40, lane);
    v16h bF[4];
#pragma unroll
    for (int n = 0; n < 4; ++n) bF[n] = load_fragB(Bs + n * 16 * 40, 40, lane);
#pragma unroll
    for (int n = 0; n < 4; ++n) acc[n] = wmma16(aF, bF[n], acc[n]);
    __syncthreads();
  }

#pragma unroll
  for (int n = 0; n < 4; ++n) {
    const int f   = fBase + n * 16 + ln;
    const float bv = bias[f];
    const int sec = f >> 10;       // 0:q 1:k 2:v (uniform per 64-wide tile col)
    const int fr  = f & (C_ - 1);
    const int h   = fr >> 6;
    const int d   = fr & (HD_ - 1);
#pragma unroll
    for (int r = 0; r < 8; ++r) {
      const int m  = mBase + wv * 16 + r + hi * 8;
      const int bb = m >> 11;            // /T_
      const int t  = m & (T_ - 1);
      const float val = acc[n][r] + bv;
      if (sec == 0) {
        Q[(((size_t)(bb * H_ + h)) * T_ + t) * HD_ + d] = (_Float16)(val * 0.125f);
      } else if (sec == 1) {
        K[(((size_t)(bb * H_ + h)) * T_ + t) * HD_ + d] = (_Float16)val;
      } else {
        Vt[(((size_t)(bb * H_ + h)) * HD_ + d) * T_ + t] = (_Float16)val;
      }
    }
  }
}

// ---------------------------------------------------------------------------
// Kernel 2: causal flash attention. grid = (T/64, B*H), 128 threads (4 waves).
// ---------------------------------------------------------------------------
__global__ __launch_bounds__(128) void attn_kernel(const _Float16* __restrict__ Q,
                                                   const _Float16* __restrict__ K,
                                                   const _Float16* __restrict__ Vt,
                                                   _Float16* __restrict__ Y) {
  __shared__ __align__(16) _Float16 Ks[32 * 72];       // [key][hd]
  __shared__ __align__(16) _Float16 Vs[64 * 40];       // [hd][key]  (V^T tile)
  __shared__ __align__(16) _Float16 Pall[4][16 * 40];  // per-wave P scratch
  const int tid = threadIdx.x, lane = tid & 31, wv = tid >> 5;
  const int hi = (lane >> 4) & 1, ln = lane & 15;
  const int bh    = blockIdx.y;
  const int qBlk  = blockIdx.x * 64;
  const int qBase = qBlk + wv * 16;
  _Float16* Pw = Pall[wv];

  const _Float16* qtile = Q + ((size_t)bh * T_ + qBase) * HD_;
  const v16h aQ0 = load_fragA(qtile, HD_, lane);
  const v16h aQ1 = load_fragA(qtile + 32, HD_, lane);

  v8f acc[4] = {};
  float mrow[8], lrow[8];
#pragma unroll
  for (int r = 0; r < 8; ++r) { mrow[r] = -1e30f; lrow[r] = 0.0f; }

  const int nChunks = (qBlk + 64) / 32;
  const _Float16* Kbh = K  + (size_t)bh * T_ * HD_;
  const _Float16* Vbh = Vt + (size_t)bh * HD_ * T_;

  const int krow = tid >> 2, kcol = (tid & 3) * 16;  // 32 rows x 64 cols
  const int vrow = tid >> 1, vcol = (tid & 1) * 16;  // 64 rows x 32 cols

  for (int c = 0; c < nChunks; ++c) {
    const int kc = c * 32;
    __syncthreads();  // previous chunk's reads done before overwrite
    {
      // async (ASYNCcnt-tracked) global -> LDS, no VGPR round-trip
      const _Float16* ks = Kbh + (size_t)(kc + krow) * HD_ + kcol;
      g2l_b128(ks,     Ks + krow * 72 + kcol);
      g2l_b128(ks + 8, Ks + krow * 72 + kcol + 8);
      const _Float16* vs = Vbh + (size_t)vrow * T_ + kc + vcol;
      g2l_b128(vs,     Vs + vrow * 40 + vcol);
      g2l_b128(vs + 8, Vs + vrow * 40 + vcol + 8);
      if (c + 1 < nChunks) {  // gfx1250 global_prefetch of next chunk
        __builtin_prefetch(Kbh + (size_t)(kc + 32 + krow) * HD_ + kcol, 0, 3);
        __builtin_prefetch(Vbh + (size_t)vrow * T_ + kc + 32 + vcol, 0, 3);
      }
    }
    wait_async();
    __syncthreads();

    // S = Q K^T (scale folded into Q): preload fragments then 4 WMMAs
    v8f s0 = {}, s1 = {};
    {
      const v16h b00 = load_fragB(Ks, 72, lane);
      const v16h b01 = load_fragB(Ks + 32, 72, lane);
      const v16h b10 = load_fragB(Ks + 16 * 72, 72, lane);
      const v16h b11 = load_fragB(Ks + 16 * 72 + 32, 72, lane);
      s0 = wmma16(aQ0, b00, s0);
      s0 = wmma16(aQ1, b01, s0);
      s1 = wmma16(aQ0, b10, s1);
      s1 = wmma16(aQ1, b11, s1);
    }

    // online softmax (row stats via 16-lane shuffle reductions)
    float p0[8], p1[8], corr[8];
#pragma unroll
    for (int r = 0; r < 8; ++r) {
      const int q = qBase + r + hi * 8;
      float a = s0[r]; if (kc + ln > q)      a = -1e30f;
      float b = s1[r]; if (kc + 16 + ln > q) b = -1e30f;
      float mx = fmaxf(a, b);
      for (int off = 8; off; off >>= 1) mx = fmaxf(mx, __shfl_xor(mx, off, 32));
      const float mnew = fmaxf(mrow[r], mx);
      const float cr = __expf(mrow[r] - mnew);
      const float e0 = __expf(a - mnew);
      const float e1 = __expf(b - mnew);
      float ps = e0 + e1;
      for (int off = 8; off; off >>= 1) ps += __shfl_xor(ps, off, 32);
      lrow[r] = lrow[r] * cr + ps;
      mrow[r] = mnew;
      corr[r] = cr;
      p0[r] = e0;
      p1[r] = e1;
    }
#pragma unroll
    for (int n = 0; n < 4; ++n)
#pragma unroll
      for (int r = 0; r < 8; ++r) acc[n][r] *= corr[r];

    // transpose P (C/D layout) -> A-fragment layout through LDS
#pragma unroll
    for (int r = 0; r < 8; ++r) {
      Pw[(r + hi * 8) * 40 + ln]      = (_Float16)p0[r];
      Pw[(r + hi * 8) * 40 + 16 + ln] = (_Float16)p1[r];
    }
    __syncthreads();
    const v16h aP = load_fragA(Pw, 40, lane);
    v16h bV[4];
#pragma unroll
    for (int n = 0; n < 4; ++n) bV[n] = load_fragB(Vs + n * 16 * 40, 40, lane);
#pragma unroll
    for (int n = 0; n < 4; ++n) acc[n] = wmma16(aP, bV[n], acc[n]);
  }

  // y -> [B,T,C] f16 workspace
  const int bb = bh >> 4, h = bh & 15;
#pragma unroll
  for (int n = 0; n < 4; ++n) {
#pragma unroll
    for (int r = 0; r < 8; ++r) {
      const int q = qBase + r + hi * 8;
      const float v = acc[n][r] / lrow[r];
      Y[((size_t)(bb * T_ + q)) * C_ + h * HD_ + n * 16 + ln] = (_Float16)v;
    }
  }
}

// ---------------------------------------------------------------------------
// Kernel 3: out = y @ w_proj^T + b_proj  (f16 A from workspace, f32 out)
// ---------------------------------------------------------------------------
__global__ __launch_bounds__(128) void proj_gemm(const _Float16* __restrict__ Yh,
                                                 const float* __restrict__ w,
                                                 const float* __restrict__ bias,
                                                 float* __restrict__ out) {
  __shared__ __align__(16) _Float16 As[64 * 40];
  __shared__ __align__(16) _Float16 Bs[64 * 40];
  const int tid = threadIdx.x, lane = tid & 31, wv = tid >> 5;
  const int hi = (lane >> 4) & 1, ln = lane & 15;
  const int mBase = blockIdx.x * 64;
  const int fBase = blockIdx.y * 64;

  v8f acc[4] = {};
  const int r0 = tid >> 3, c4 = (tid & 7) * 4;
  const int arow = tid >> 1, acol = (tid & 1) * 16;

  for (int kc = 0; kc < C_; kc += 32) {
    // A tile: pure f16 copy -> async global->LDS
    {
      const _Float16* asrc = Yh + (size_t)(mBase + arow) * C_ + kc + acol;
      g2l_b128(asrc,     As + arow * 40 + acol);
      g2l_b128(asrc + 8, As + arow * 40 + acol + 8);
    }
    // B tile: f32 -> f16 conversion, loads first then converts
    float4 bv[4];
#pragma unroll
    for (int rr = 0; rr < 4; ++rr) {
      const int row = rr * 16 + r0;
      bv[rr] = *(const float4*)(w + (size_t)(fBase + row) * C_ + kc + c4);
    }
#pragma unroll
    for (int rr = 0; rr < 4; ++rr) {
      const int row = rr * 16 + r0;
      v4h bh4 = {(_Float16)bv[rr].x, (_Float16)bv[rr].y, (_Float16)bv[rr].z, (_Float16)bv[rr].w};
      *(v4h*)(Bs + row * 40 + c4) = bh4;
    }
    wait_async();
    __syncthreads();
    const v16h aF = load_fragA(As + wv * 16 * 40, 40, lane);
    v16h bF[4];
#pragma unroll
    for (int n = 0; n < 4; ++n) bF[n] = load_fragB(Bs + n * 16 * 40, 40, lane);
#pragma unroll
    for (int n = 0; n < 4; ++n) acc[n] = wmma16(aF, bF[n], acc[n]);
    __syncthreads();
  }

#pragma unroll
  for (int n = 0; n < 4; ++n) {
    const int f = fBase + n * 16 + ln;
    const float bv = bias[f];
#pragma unroll
    for (int r = 0; r < 8; ++r) {
      const int m = mBase + wv * 16 + r + hi * 8;
      out[(size_t)m * C_ + f] = acc[n][r] + bv;
    }
  }
}

// ---------------------------------------------------------------------------
extern "C" void kernel_launch(void* const* d_in, const int* in_sizes, int n_in,
                              void* d_out, int out_size, void* d_ws, size_t ws_size,
                              hipStream_t stream) {
  const float* x      = (const float*)d_in[0];
  const float* w_attn = (const float*)d_in[1];
  const float* b_attn = (const float*)d_in[2];
  const float* w_proj = (const float*)d_in[3];
  const float* b_proj = (const float*)d_in[4];
  float* out = (float*)d_out;

  _Float16* ws = (_Float16*)d_ws;
  const size_t seg = (size_t)B_ * H_ * T_ * HD_;  // 8,388,608 halves = 16 MB
  _Float16* Q  = ws;
  _Float16* K  = ws + seg;
  _Float16* Vt = ws + 2 * seg;
  _Float16* Y  = ws + 3 * seg;  // [B,T,C] f16

  qkv_gemm<<<dim3(M_ / 64, N3_ / 64), 128, 0, stream>>>(x, w_attn, b_attn, Q, K, Vt);
  attn_kernel<<<dim3(T_ / 64, B_ * H_), 128, 0, stream>>>(Q, K, Vt, Y);
  proj_gemm<<<dim3(M_ / 64, C_ / 64), 128, 0, stream>>>(Y, w_proj, b_proj, out);
}
